// SelfAttention_82789789597918
// MI455X (gfx1250) — compile-verified, matrix operand order checked by reading, not run on hardware
//
#include <hip/hip_runtime.h>

// ---------------------------------------------------------------------------
// Types
// ---------------------------------------------------------------------------
typedef __attribute__((ext_vector_type(16))) __bf16 v16bf;
typedef __attribute__((ext_vector_type(8)))  __bf16 v8bf;
typedef __attribute__((ext_vector_type(8)))  float  v8f;
typedef __attribute__((ext_vector_type(4)))  unsigned int u32x4;

#define B_  4
#define S_  2048
#define E_  1024
#define H_  4
#define D_  256

#define BM 128
#define BN 64
#define BK 32

__device__ __forceinline__ v8f vzero8() {
  v8f z;
#pragma unroll
  for (int i = 0; i < 8; ++i) z[i] = 0.0f;
  return z;
}

__device__ __forceinline__ v16bf combine16(v8bf lo, v8bf hi) {
  v16bf r;
#pragma unroll
  for (int i = 0; i < 8; ++i) { r[i] = lo[i]; r[8 + i] = hi[i]; }
  return r;
}

// WMMA A/B fragment (16 rows x 32 k bf16) from LDS.
// ISA 7.12.2 lane layout: lanes 0-15: K=[0..7],[16..23]; lanes 16-31:
// K=[8..15],[24..31]; row = lane & 15.
__device__ __forceinline__ v16bf load_frag(const __bf16* base, int ld, int lane) {
  int r  = lane & 15;
  int kh = lane >> 4;
  const __bf16* p = base + r * ld + kh * 8;
  v8bf lo = *(const v8bf*)(p);
  v8bf hi = *(const v8bf*)(p + 16);
  return combine16(lo, hi);
}

__device__ __forceinline__ v8f wmma_bf16(v16bf a, v16bf b, v8f c) {
  return __builtin_amdgcn_wmma_f32_16x16x32_bf16(
      false, a, false, b, (short)0, c, false, false);
}

// CDNA5 async global->LDS DMA (16B per lane), tracked by ASYNCcnt.
__device__ __forceinline__ void async_load_b128(const void* lds_dst, const void* gsrc) {
  asm volatile("global_load_async_to_lds_b128 %0, %1, off"
               :: "v"((unsigned)(uintptr_t)lds_dst), "v"(gsrc)
               : "memory");
}
__device__ __forceinline__ void wait_async0() {
  asm volatile("s_wait_asynccnt 0" ::: "memory");
}

// ---------------------------------------------------------------------------
// Stage 1a: fp32 -> bf16 (grid-stride)
// ---------------------------------------------------------------------------
__global__ __launch_bounds__(256) void cvt_f32_bf16(const float* __restrict__ in,
                                                    __bf16* __restrict__ out,
                                                    long n) {
  long i = (long)blockIdx.x * 256 + threadIdx.x;
  long stride = (long)gridDim.x * 256;
  for (; i < n; i += stride) out[i] = (__bf16)in[i];
}

// ---------------------------------------------------------------------------
// Stage 1b: fp32 [K,N] -> bf16 [N,K] transposed (batched; K,N powers of two).
// ---------------------------------------------------------------------------
__global__ __launch_bounds__(256) void cvt_transpose_bf16(const float* __restrict__ in,
                                                          __bf16* __restrict__ out,
                                                          int logK, int logN, long total) {
  long i = (long)blockIdx.x * 256 + threadIdx.x;
  long stride = (long)gridDim.x * 256;
  const int logMat = logK + logN;
  for (; i < total; i += stride) {
    long mat = i >> logMat;
    long rem = i & ((1L << logMat) - 1);
    int n = (int)(rem >> logK);
    int k = (int)(rem & ((1 << logK) - 1));
    out[i] = (__bf16)in[(mat << logMat) + ((long)k << logN) + n];
  }
}

// ---------------------------------------------------------------------------
// Stage 2: QKV projection GEMM (async double-buffered LDS, 1 barrier/step).
//   A  : x_bf16 [B*S, E] row-major
//   Wt : [H, D, E] bf16 (pre-transposed weight: row n holds all k)
//   Out: TRANS==0 -> [B,H,S,D] ; TRANS==1 -> [B,H,D,S]
// ---------------------------------------------------------------------------
template <int TRANS>
__global__ __launch_bounds__(256) void qkv_gemm(const __bf16* __restrict__ A,
                                                const __bf16* __restrict__ Wt,
                                                __bf16* __restrict__ Out) {
  __shared__ __align__(16) __bf16 smem[16384];        // 32 KB
  __bf16* As = smem;                                  // 2 x 128*32
  __bf16* Bs = smem + 2 * (BM * BK);                  // 2 x 64*32

  const int tid = threadIdx.x;
  const int lane = tid & 31, wave = tid >> 5;
  const int waveM = wave >> 1, waveN = wave & 1;
  const int tileM = blockIdx.x * BM;
  const int h     = blockIdx.y >> 2;
  const int tileN = (blockIdx.y & 3) * BN;
  const __bf16* Wh = Wt + (long)h * D_ * E_;

  const int ar = tid >> 1, ac = (tid & 1) * 16;       // A: 128x32, 32B/thread
  const int bn = tid >> 2, bk = (tid & 3) * 8;        // B: 64x32, 16B/thread

  auto issue = [&](int k0, int p) {
    __bf16* Ad = As + p * (BM * BK) + ar * BK + ac;
    const __bf16* Asrc = A + (long)(tileM + ar) * E_ + k0 + ac;
    async_load_b128(Ad, Asrc);
    async_load_b128(Ad + 8, Asrc + 8);
    async_load_b128(Bs + p * (BN * BK) + bn * BK + bk,
                    Wh + (long)(tileN + bn) * E_ + k0 + bk);
  };

  v8f acc[2][2];
#pragma unroll
  for (int i = 0; i < 2; ++i)
#pragma unroll
    for (int j = 0; j < 2; ++j) acc[i][j] = vzero8();

  issue(0, 0);
  int p = 0;
  for (int k0 = 0; k0 < E_; k0 += BK) {
    wait_async0();
    __syncthreads();                      // tile p ready; buffer p^1 free
    if (k0 + BK < E_) issue(k0 + BK, p ^ 1);
    const __bf16* Ab = As + p * (BM * BK);
    const __bf16* Bb = Bs + p * (BN * BK);
    v16bf af[2], bf[2];
#pragma unroll
    for (int i = 0; i < 2; ++i)
      af[i] = load_frag(Ab + (waveM * 32 + i * 16) * BK, BK, lane);
#pragma unroll
    for (int j = 0; j < 2; ++j)
      bf[j] = load_frag(Bb + (waveN * 32 + j * 16) * BK, BK, lane);
#pragma unroll
    for (int i = 0; i < 2; ++i)
#pragma unroll
      for (int j = 0; j < 2; ++j) acc[i][j] = wmma_bf16(af[i], bf[j], acc[i][j]);
    p ^= 1;
  }
  __syncthreads();                        // all compute done before smem reuse

  // Epilogue: stage wave's 32x32 tile in wave-private LDS (same-wave DS ops
  // are ordered -> no barrier), then coalesced b128 stores.
  __bf16* Wst = smem + wave * 1024;
#pragma unroll
  for (int i = 0; i < 2; ++i) {
#pragma unroll
    for (int j = 0; j < 2; ++j) {
#pragma unroll
      for (int e = 0; e < 8; ++e) {
        int r = i * 16 + e + ((lane >> 4) << 3);
        int c = j * 16 + (lane & 15);
        if (TRANS == 0) Wst[r * 32 + c] = (__bf16)acc[i][j][e];
        else            Wst[c * 32 + r] = (__bf16)acc[i][j][e];
      }
    }
  }
  const u32x4* src = (const u32x4*)(Wst + lane * 32);
  u32x4 d0 = src[0], d1 = src[1], d2 = src[2], d3 = src[3];
  const int gb = tileM >> 11;
  __bf16* dp;
  if (TRANS == 0) {
    int s = (tileM + waveM * 32 + lane) & (S_ - 1);
    dp = Out + ((long)(gb * H_ + h) * S_ + s) * D_ + tileN + waveN * 32;
  } else {
    int col = tileN + waveN * 32 + lane;
    int s0 = (tileM + waveM * 32) & (S_ - 1);
    dp = Out + ((long)(gb * H_ + h) * D_ + col) * S_ + s0;
  }
  u32x4* d = (u32x4*)dp;
  d[0] = d0; d[1] = d1; d[2] = d2; d[3] = d3;
}

// ---------------------------------------------------------------------------
// Stage 3: flash attention (causal, online softmax).
// Double-buffered async K/V tiles, one barrier per 32-key step.
//   Q,K : [B,H,S,D] bf16 ; Vt : [B,H,D,S] bf16 ; Outp : [B,S,E] bf16
// ---------------------------------------------------------------------------
__global__ __launch_bounds__(256) void flash_attn(const __bf16* __restrict__ Q,
                                                  const __bf16* __restrict__ Kg,
                                                  const __bf16* __restrict__ Vt,
                                                  __bf16* __restrict__ Outp) {
  __shared__ __align__(16) __bf16 Ks[2 * 32 * D_];  // [p][key][d]  32 KB
  __shared__ __align__(16) __bf16 Vs[2 * D_ * 32];  // [p][d][key]  32 KB
  __shared__ __align__(16) __bf16 Ps[8][16 * 32];   // per-wave      8 KB

  const int tid = threadIdx.x;
  const int lane = tid & 31, wave = tid >> 5;
  const int bh = blockIdx.y;
  const int b = bh >> 2, h = bh & 3;
  const int q0 = blockIdx.x * 128;
  const int qrow = q0 + wave * 16;

  const __bf16* Qbh = Q  + (long)bh * S_ * D_;
  const __bf16* Kbh = Kg + (long)bh * S_ * D_;
  const __bf16* Vbh = Vt + (long)bh * D_ * S_;

  // Q fragments resident in registers (8 x k=32 chunks over D=256).
  v16bf qf[8];
  {
    int r = lane & 15, kh = lane >> 4;
    const __bf16* qp = Qbh + (long)(qrow + r) * D_;
#pragma unroll
    for (int dc = 0; dc < 8; ++dc) {
      v8bf lo = *(const v8bf*)(qp + dc * 32 + kh * 8);
      v8bf hi = *(const v8bf*)(qp + dc * 32 + 16 + kh * 8);
      qf[dc] = combine16(lo, hi);
    }
  }

  v8f o[16];
#pragma unroll
  for (int dc = 0; dc < 16; ++dc) o[dc] = vzero8();
  float mrow[8], lrow[8];
#pragma unroll
  for (int e = 0; e < 8; ++e) { mrow[e] = -3.0e38f; lrow[e] = 0.0f; }

  const float scale = 0.03125f;  // 1/sqrt(E)
  const int kend = q0 + 128;     // causal bound
  const int kkey = tid >> 3, kd0 = (tid & 7) * 32;

  auto issueKV = [&](int ks, int p) {
    const __bf16* srcK = Kbh + (long)(ks + kkey) * D_ + kd0;
    __bf16* dstK = Ks + p * (32 * D_) + kkey * D_ + kd0;
#pragma unroll
    for (int j = 0; j < 4; ++j) async_load_b128(dstK + j * 8, srcK + j * 8);
    const __bf16* srcV = Vbh + (long)tid * S_ + ks;
    __bf16* dstV = Vs + p * (D_ * 32) + tid * 32;
#pragma unroll
    for (int j = 0; j < 4; ++j) async_load_b128(dstV + j * 8, srcV + j * 8);
  };

  issueKV(0, 0);
  int p = 0;
  for (int ks = 0; ks < kend; ks += 32) {
    wait_async0();
    __syncthreads();                      // tiles p ready; buffers p^1 free
    if (ks + 32 < kend) issueKV(ks + 32, p ^ 1);
    const __bf16* Kbuf = Ks + p * (32 * D_);
    const __bf16* Vbuf = Vs + p * (D_ * 32);

    // Scores: two 16x16 tiles (32 keys), 8 wmma each over D=256.
    v8f sc[2];
#pragma unroll
    for (int sub = 0; sub < 2; ++sub) {
      v8f c = vzero8();
#pragma unroll
      for (int dc = 0; dc < 8; ++dc) {
        v16bf bf = load_frag(Kbuf + (sub * 16) * D_ + dc * 32, D_, lane);
        c = wmma_bf16(qf[dc], bf, c);
      }
      sc[sub] = c;
    }

    // Online softmax (row reductions across each 16-lane half).
    const int colg0 = ks + (lane & 15);
    const int rowg0 = qrow + ((lane >> 4) << 3);
#pragma unroll
    for (int e = 0; e < 8; ++e) {
      int rowg = rowg0 + e;
      float s0 = sc[0][e] * scale; if (colg0      > rowg) s0 = -1.0e9f;
      float s1 = sc[1][e] * scale; if (colg0 + 16 > rowg) s1 = -1.0e9f;
      float mx = fmaxf(s0, s1);
#pragma unroll
      for (int off = 8; off >= 1; off >>= 1) mx = fmaxf(mx, __shfl_xor(mx, off, 32));
      float mnew = fmaxf(mrow[e], mx);
      float corr = __expf(mrow[e] - mnew);
      float p0 = __expf(s0 - mnew);
      float p1 = __expf(s1 - mnew);
      float psum = p0 + p1;
#pragma unroll
      for (int off = 8; off >= 1; off >>= 1) psum += __shfl_xor(psum, off, 32);
      lrow[e] = lrow[e] * corr + psum;
      mrow[e] = mnew;
#pragma unroll
      for (int dc = 0; dc < 16; ++dc) o[dc][e] *= corr;
      int prow = e + ((lane >> 4) << 3);
      Ps[wave][prow * 32 + (lane & 15)]      = (__bf16)p0;
      Ps[wave][prow * 32 + 16 + (lane & 15)] = (__bf16)p1;
    }

    // P as A operand; PV accumulate (16 d-tiles, K=32 keys).
    v16bf pf = load_frag(&Ps[wave][0], 32, lane);
#pragma unroll
    for (int dc = 0; dc < 16; ++dc) {
      v16bf bf = load_frag(Vbuf + (dc * 16) * 32, 32, lane);
      o[dc] = wmma_bf16(pf, bf, o[dc]);
    }
    p ^= 1;
  }

  // Normalize; stage 16x32 chunks through wave-private Ps, store coalesced.
  float inv[8];
#pragma unroll
  for (int e = 0; e < 8; ++e) inv[e] = 1.0f / lrow[e];
  const int prow0 = (lane >> 4) << 3;
  const int cr = lane >> 1, cc0 = (lane & 1) * 16;
  __bf16* rowbase = Outp + ((long)(b * S_ + qrow)) * E_ + h * D_;
#pragma unroll
  for (int g = 0; g < 8; ++g) {
#pragma unroll
    for (int half = 0; half < 2; ++half) {
      int dc = g * 2 + half;
#pragma unroll
      for (int e = 0; e < 8; ++e)
        Ps[wave][(prow0 + e) * 32 + half * 16 + (lane & 15)] =
            (__bf16)(o[dc][e] * inv[e]);
    }
    const u32x4* sp = (const u32x4*)(&Ps[wave][cr * 32 + cc0]);
    u32x4 v0 = sp[0], v1 = sp[1];
    u32x4* dp = (u32x4*)(rowbase + (long)cr * E_ + g * 32 + cc0);
    dp[0] = v0; dp[1] = v1;
  }
}

// ---------------------------------------------------------------------------
// Stage 4: output projection GEMM + residual -> fp32 d_out.
//   A: attn bf16 [B*S, E]; WoT: out_w transposed bf16 [N=E][K=E]; resid: x f32
// ---------------------------------------------------------------------------
__global__ __launch_bounds__(256) void proj_gemm(const __bf16* __restrict__ A,
                                                 const __bf16* __restrict__ WoT,
                                                 const float* __restrict__ resid,
                                                 float* __restrict__ Out) {
  __shared__ __align__(16) __bf16 smem[16384];        // 32 KB
  __bf16* As = smem;
  __bf16* Bs = smem + 2 * (BM * BK);

  const int tid = threadIdx.x;
  const int lane = tid & 31, wave = tid >> 5;
  const int waveM = wave >> 1, waveN = wave & 1;
  const int tileM = blockIdx.x * BM;
  const int tileN = blockIdx.y * BN;

  const int ar = tid >> 1, ac = (tid & 1) * 16;
  const int bn = tid >> 2, bk = (tid & 3) * 8;

  auto issue = [&](int k0, int p) {
    __bf16* Ad = As + p * (BM * BK) + ar * BK + ac;
    const __bf16* Asrc = A + (long)(tileM + ar) * E_ + k0 + ac;
    async_load_b128(Ad, Asrc);
    async_load_b128(Ad + 8, Asrc + 8);
    async_load_b128(Bs + p * (BN * BK) + bn * BK + bk,
                    WoT + (long)(tileN + bn) * E_ + k0 + bk);
  };

  v8f acc[2][2];
#pragma unroll
  for (int i = 0; i < 2; ++i)
#pragma unroll
    for (int j = 0; j < 2; ++j) acc[i][j] = vzero8();

  issue(0, 0);
  int p = 0;
  for (int k0 = 0; k0 < E_; k0 += BK) {
    wait_async0();
    __syncthreads();
    if (k0 + BK < E_) issue(k0 + BK, p ^ 1);
    const __bf16* Ab = As + p * (BM * BK);
    const __bf16* Bb = Bs + p * (BN * BK);
    v16bf af[2], bf[2];
#pragma unroll
    for (int i = 0; i < 2; ++i)
      af[i] = load_frag(Ab + (waveM * 32 + i * 16) * BK, BK, lane);
#pragma unroll
    for (int j = 0; j < 2; ++j)
      bf[j] = load_frag(Bb + (waveN * 32 + j * 16) * BK, BK, lane);
#pragma unroll
    for (int i = 0; i < 2; ++i)
#pragma unroll
      for (int j = 0; j < 2; ++j) acc[i][j] = wmma_bf16(af[i], bf[j], acc[i][j]);
    p ^= 1;
  }
  __syncthreads();

  // Epilogue: stage f32 tile per wave (4KB x 8 waves = full smem), then
  // coalesced float4 loads/stores with residual add.
  float* Wst = (float*)smem + wave * 1024;
#pragma unroll
  for (int i = 0; i < 2; ++i)
#pragma unroll
    for (int j = 0; j < 2; ++j)
#pragma unroll
      for (int e = 0; e < 8; ++e) {
        int r = i * 16 + e + ((lane >> 4) << 3);
        int c = j * 16 + (lane & 15);
        Wst[r * 32 + c] = acc[i][j][e];
      }
  const int row = tileM + waveM * 32 + lane;
  float* dp = Out + (long)row * E_ + tileN + waveN * 32;
  const float* rp = resid + (long)row * E_ + tileN + waveN * 32;
  const float4* sp = (const float4*)(Wst + lane * 32);
#pragma unroll
  for (int q = 0; q < 8; ++q) {
    float4 a = sp[q];
    float4 rr = ((const float4*)rp)[q];
    float4 ov;
    ov.x = a.x + rr.x; ov.y = a.y + rr.y; ov.z = a.z + rr.z; ov.w = a.w + rr.w;
    ((float4*)dp)[q] = ov;
  }
}

// ---------------------------------------------------------------------------
// Stage 5: in-place LayerNorm over d_out rows.
// ---------------------------------------------------------------------------
__global__ __launch_bounds__(256) void layernorm_kernel(float* __restrict__ y,
                                                        const float* __restrict__ g,
                                                        const float* __restrict__ bt) {
  __shared__ float red[256];
  const int tid = threadIdx.x;
  float* p = y + (long)blockIdx.x * E_;

  float4 v = ((const float4*)p)[tid];
  red[tid] = v.x + v.y + v.z + v.w;
  __syncthreads();
#pragma unroll
  for (int st = 128; st > 0; st >>= 1) {
    if (tid < st) red[tid] += red[tid + st];
    __syncthreads();
  }
  float mean = red[0] * (1.0f / E_);
  __syncthreads();

  float dx = v.x - mean, dy = v.y - mean, dz = v.z - mean, dw = v.w - mean;
  red[tid] = dx * dx + dy * dy + dz * dz + dw * dw;
  __syncthreads();
#pragma unroll
  for (int st = 128; st > 0; st >>= 1) {
    if (tid < st) red[tid] += red[tid + st];
    __syncthreads();
  }
  float rs = rsqrtf(red[0] * (1.0f / E_) + 1e-6f);

  float4 gg = ((const float4*)g)[tid];
  float4 bb = ((const float4*)bt)[tid];
  float4 ov;
  ov.x = dx * rs * gg.x + bb.x;
  ov.y = dy * rs * gg.y + bb.y;
  ov.z = dz * rs * gg.z + bb.z;
  ov.w = dw * rs * gg.w + bb.w;
  ((float4*)p)[tid] = ov;
}

// ---------------------------------------------------------------------------
// Launch
// ---------------------------------------------------------------------------
extern "C" void kernel_launch(void* const* d_in, const int* in_sizes, int n_in,
                              void* d_out, int out_size, void* d_ws, size_t ws_size,
                              hipStream_t stream) {
  (void)in_sizes; (void)n_in; (void)out_size; (void)ws_size;

  const float* x     = (const float*)d_in[0]; // [B,S,E]
  const float* qkv_w = (const float*)d_in[1]; // [3,H,E,D]
  const float* out_w = (const float*)d_in[2]; // [E,E]
  const float* gamma = (const float*)d_in[3];
  const float* beta  = (const float*)d_in[4];
  float* out = (float*)d_out;

  const long nX  = (long)B_ * S_ * E_;        // 8388608
  const long nW3 = 3L * H_ * E_ * D_;         // 3145728
  const long nWo = (long)E_ * E_;             // 1048576
  const long nQ  = (long)B_ * H_ * S_ * D_;   // 8388608

  char* ws = (char*)d_ws;
  __bf16* xb    = (__bf16*)(ws);
  __bf16* wqkvT = (__bf16*)(ws + 2 * nX);                       // [3,H,D,E]
  __bf16* woT   = (__bf16*)(ws + 2 * (nX + nW3));               // [E,E] (NT)
  __bf16* Qb    = (__bf16*)(ws + 2 * (nX + nW3 + nWo));
  __bf16* Kb    = (__bf16*)(ws + 2 * (nX + nW3 + nWo + nQ));
  __bf16* Vtb   = (__bf16*)(ws + 2 * (nX + nW3 + nWo + 2 * nQ));
  __bf16* attn  = (__bf16*)(ws + 2 * (nX + nW3 + nWo + 3 * nQ));

  // 1) conversions (+ weight transposes for contiguous WMMA B operands)
  cvt_f32_bf16<<<4096, 256, 0, stream>>>(x, xb, nX);
  cvt_transpose_bf16<<<2048, 256, 0, stream>>>(qkv_w, wqkvT, 10, 8, nW3);
  cvt_transpose_bf16<<<1024, 256, 0, stream>>>(out_w, woT, 10, 10, nWo);

  // 2) QKV projections (V stored transposed [B,H,D,S])
  dim3 ggrid(64, 16);
  qkv_gemm<0><<<ggrid, 256, 0, stream>>>(xb, wqkvT,                  Qb);
  qkv_gemm<0><<<ggrid, 256, 0, stream>>>(xb, wqkvT + (long)H_*D_*E_, Kb);
  qkv_gemm<1><<<ggrid, 256, 0, stream>>>(xb, wqkvT + 2L*H_*D_*E_,    Vtb);

  // 3) Flash attention -> attn [B,S,E] bf16
  flash_attn<<<dim3(S_ / 128, B_ * H_), 256, 0, stream>>>(Qb, Kb, Vtb, attn);

  // 4) Output projection + residual -> d_out (fp32)
  proj_gemm<<<dim3(64, 16), 256, 0, stream>>>(attn, woT, x, out);

  // 5) In-place LayerNorm on d_out
  layernorm_kernel<<<B_ * S_, 256, 0, stream>>>(out, gamma, beta);
}
